// MultiheadAttention_26027501814334
// MI455X (gfx1250) — matrix operand order repaired and checked by reading, not compiled, not run on hardware
//
#include <hip/hip_runtime.h>
#include <hip/hip_bf16.h>
#include <math.h>

// Problem constants (from reference)
#define SQL   2048
#define SKVL  2048
#define BATCH 4
#define EM    1024
#define NHEAD 16
#define HEADD 64

typedef __bf16 bf16_t;
typedef __attribute__((ext_vector_type(16))) __bf16 v16bf;
typedef __attribute__((ext_vector_type(8)))  __bf16 v8bf;
typedef __attribute__((ext_vector_type(8)))  float  v8f;

// ---- fragment load helpers -------------------------------------------------
// A-matrix (16x32 bf16) layout: lane L holds row M=L%16; kbase=(L/16)*16;
//   a[0..7]  = A[M, kbase..kbase+7], a[8..15] = A[M, 8+kbase .. 8+kbase+7]
// B-matrix (32x16 bf16) layout: lane L holds col N=L%16; kbase=(L/16)*16;
//   b[i] = B[kbase+i, N]  (contiguous along K for our operands)

__device__ inline v16bf ld_a_from_f32(const float* __restrict__ row, int akb) {
    v16bf a;
#pragma unroll
    for (int i = 0; i < 8; ++i) {
        a[i]     = (__bf16)row[akb + i];
        a[i + 8] = (__bf16)row[8 + akb + i];
    }
    return a;
}

__device__ inline v16bf ld_a_from_bf16(const bf16_t* __restrict__ row, int akb) {
    const v8bf* p0 = (const v8bf*)(row + akb);
    const v8bf* p1 = (const v8bf*)(row + 8 + akb);
    v8bf lo = *p0, hi = *p1;
    v16bf a;
#pragma unroll
    for (int i = 0; i < 8; ++i) { a[i] = lo[i]; a[i + 8] = hi[i]; }
    return a;
}

__device__ inline v16bf ld_b_from_f32(const float* __restrict__ p) {
    v16bf b;
#pragma unroll
    for (int i = 0; i < 16; ++i) b[i] = (__bf16)p[i];
    return b;
}

__device__ inline v16bf ld_b_from_bf16(const bf16_t* __restrict__ p) {
    const v8bf* q = (const v8bf*)p;
    v8bf lo = q[0], hi = q[1];
    v16bf b;
#pragma unroll
    for (int i = 0; i < 8; ++i) { b[i] = lo[i]; b[i + 8] = hi[i]; }
    return b;
}

// ---- Kernel 1: QKV projection (x @ W^T + b), bf16 output -------------------
// grid.x = 256 (wave tiles / 8), grid.y = 3 (Q,K,V), block = 256 (8 waves)
// Each wave: 64x64 output tile. Q,K -> [B,H,S,D]; V -> [B,H,D,S] (transposed).
__global__ __launch_bounds__(256)
void mha_qkv_proj(const float* __restrict__ q_em,
                  const float* __restrict__ kv_em,
                  const float* __restrict__ Wq, const float* __restrict__ bq,
                  const float* __restrict__ Wk, const float* __restrict__ bk,
                  const float* __restrict__ Wv, const float* __restrict__ bv,
                  bf16_t* __restrict__ Qo, bf16_t* __restrict__ Ko,
                  bf16_t* __restrict__ Vo)
{
    const int lane  = threadIdx.x & 31;
    const int wid   = threadIdx.x >> 5;
    const int which = blockIdx.y;                 // 0=Q 1=K 2=V (uniform)
    const int wtile = blockIdx.x * 8 + wid;       // 0..2047

    const int b     = wtile >> 9;                 // 512 tiles per batch
    const int t     = wtile & 511;
    const int stile = t >> 4;                     // 32 row-tiles (64 rows)
    const int ntile = t & 15;                     // 16 col-tiles (64 cols)

    const float* x    = (which == 0) ? q_em : kv_em;
    const float* W    = (which == 0) ? Wq : (which == 1) ? Wk : Wv;
    const float* bias = (which == 0) ? bq : (which == 1) ? bk : bv;

    const int s_base = stile * 64;
    const int n_base = ntile * 64;
    const int mlane  = lane & 15;
    const int hilo   = lane >> 4;
    const int akb    = hilo * 16;
    const int bkb    = hilo * 16;

    v8f acc[4][4];
#pragma unroll
    for (int i = 0; i < 4; ++i)
#pragma unroll
        for (int j = 0; j < 4; ++j)
#pragma unroll
            for (int r = 0; r < 8; ++r) acc[i][j][r] = 0.0f;

    for (int k = 0; k < EM; k += 32) {
        v16bf afrag[4];
#pragma unroll
        for (int mt = 0; mt < 4; ++mt) {
            const int s = s_base + mt * 16 + mlane;
            afrag[mt] = ld_a_from_f32(x + ((size_t)s * BATCH + b) * EM + k, akb);
        }
        v16bf bfrag[4];
#pragma unroll
        for (int nt = 0; nt < 4; ++nt) {
            const int e = n_base + nt * 16 + mlane;
            bfrag[nt] = ld_b_from_f32(W + (size_t)e * EM + k + bkb);
        }
#pragma unroll
        for (int mt = 0; mt < 4; ++mt)
#pragma unroll
            for (int nt = 0; nt < 4; ++nt)
                acc[mt][nt] = __builtin_amdgcn_wmma_f32_16x16x32_bf16(
                    false, afrag[mt], false, bfrag[nt], (short)0, acc[mt][nt],
                    false, false);
    }

    // epilogue: bias add, bf16 convert, head-split store
#pragma unroll
    for (int mt = 0; mt < 4; ++mt) {
#pragma unroll
        for (int nt = 0; nt < 4; ++nt) {
            const int e  = n_base + nt * 16 + mlane;
            const int h  = e >> 6;
            const int d  = e & 63;
            const float be = bias[e];
#pragma unroll
            for (int r = 0; r < 8; ++r) {
                const int s   = s_base + mt * 16 + r + hilo * 8;
                const float v = acc[mt][nt][r] + be;
                if (which == 2) {
                    // V transposed: [B,H,D,S]
                    Vo[(((size_t)b * NHEAD + h) * HEADD + d) * SKVL + s] = (__bf16)v;
                } else if (which == 0) {
                    Qo[(((size_t)b * NHEAD + h) * SQL + s) * HEADD + d] = (__bf16)v;
                } else {
                    Ko[(((size_t)b * NHEAD + h) * SKVL + s) * HEADD + d] = (__bf16)v;
                }
            }
        }
    }
}

// ---- Kernel 2: flash attention --------------------------------------------
// grid.x = 512, block = 256 (8 waves). One wave: 32 query rows of one (b,h):
// two 16-row sub-tiles (u=0,1) sharing every K and V fragment -> 16 WMMAs per
// 8 global fragment loads per 32-wide kv step.
__global__ __launch_bounds__(256)
void mha_flash_attn(const bf16_t* __restrict__ Qm, const bf16_t* __restrict__ Km,
                    const bf16_t* __restrict__ Vt, float* __restrict__ out)
{
    __shared__ __bf16 pbuf[8][2][16 * 32];       // per-wave, per-subtile P

    const int lane  = threadIdx.x & 31;
    const int wid   = threadIdx.x >> 5;
    const int mlane = lane & 15;
    const int hilo  = lane >> 4;
    const int akb   = hilo * 16;

    const int wtile = blockIdx.x * 8 + wid;      // 0..4095
    const int qtile = wtile & 63;                // SQ/32 = 64
    const int h     = (wtile >> 6) & 15;
    const int b     = wtile >> 10;

    const bf16_t* Qb = Qm + ((size_t)b * NHEAD + h) * SQL  * HEADD;
    const bf16_t* Kb = Km + ((size_t)b * NHEAD + h) * SKVL * HEADD;
    const bf16_t* Vb = Vt + ((size_t)b * NHEAD + h) * HEADD * SKVL;

    const int q_base = qtile * 32;

    // Q fragments: [subtile u][d-chunk c]
    v16bf qf[2][2];
#pragma unroll
    for (int u = 0; u < 2; ++u) {
        const bf16_t* row = Qb + (size_t)(q_base + u * 16 + mlane) * HEADD;
        qf[u][0] = ld_a_from_bf16(row, akb);
        qf[u][1] = ld_a_from_bf16(row + 32, akb);
    }

    float m[2][8], l[2][8];
    v8f oacc[2][4];
#pragma unroll
    for (int u = 0; u < 2; ++u) {
#pragma unroll
        for (int r = 0; r < 8; ++r) { m[u][r] = -3.0e38f; l[u][r] = 0.0f; }
#pragma unroll
        for (int t = 0; t < 4; ++t)
#pragma unroll
            for (int r = 0; r < 8; ++r) oacc[u][t][r] = 0.0f;
    }

    const float scale = 0.125f;                  // 1/sqrt(64)

    for (int kv = 0; kv < SKVL; kv += 32) {
        // ---- K fragments (shared by both q sub-tiles) ----------------------
        v16bf kf[2][2];
#pragma unroll
        for (int t = 0; t < 2; ++t)
#pragma unroll
            for (int c = 0; c < 2; ++c) {
                // B[k=d, n=kv] = K[kv_base + n, d]; contiguous along d
                const bf16_t* row =
                    Kb + (size_t)(kv + t * 16 + mlane) * HEADD + c * 32 + hilo * 16;
                kf[t][c] = ld_b_from_bf16(row);
            }

        // ---- S = Q K^T : 32x32 in four 16x16 accumulators ------------------
        v8f sacc[2][2];
#pragma unroll
        for (int u = 0; u < 2; ++u)
#pragma unroll
            for (int t = 0; t < 2; ++t) {
#pragma unroll
                for (int r = 0; r < 8; ++r) sacc[u][t][r] = 0.0f;
#pragma unroll
                for (int c = 0; c < 2; ++c)
                    sacc[u][t] = __builtin_amdgcn_wmma_f32_16x16x32_bf16(
                        false, qf[u][c], false, kf[t][c], (short)0, sacc[u][t],
                        false, false);
            }

        // ---- online softmax per sub-tile -----------------------------------
        asm volatile("s_wait_dscnt 0" ::: "memory");   // WAR vs last iter's reads
#pragma unroll
        for (int u = 0; u < 2; ++u) {
            __bf16* pl = pbuf[wid][u];
            float csc[8];
#pragma unroll
            for (int r = 0; r < 8; ++r) {
                sacc[u][0][r] *= scale;
                sacc[u][1][r] *= scale;
                float v = fmaxf(sacc[u][0][r], sacc[u][1][r]);
                v = fmaxf(v, __shfl_xor(v, 1, 32));
                v = fmaxf(v, __shfl_xor(v, 2, 32));
                v = fmaxf(v, __shfl_xor(v, 4, 32));
                v = fmaxf(v, __shfl_xor(v, 8, 32));
                const float mn = fmaxf(m[u][r], v);
                csc[r]  = __expf(m[u][r] - mn);
                m[u][r] = mn;
            }
#pragma unroll
            for (int r = 0; r < 8; ++r) {
                const int row = r + hilo * 8;
                const float p0 = __expf(sacc[u][0][r] - m[u][r]);
                const float p1 = __expf(sacc[u][1][r] - m[u][r]);
                pl[row * 32 + mlane]      = (__bf16)p0;
                pl[row * 32 + 16 + mlane] = (__bf16)p1;
                float v = p0 + p1;
                v += __shfl_xor(v, 1, 32);
                v += __shfl_xor(v, 2, 32);
                v += __shfl_xor(v, 4, 32);
                v += __shfl_xor(v, 8, 32);
                l[u][r] = l[u][r] * csc[r] + v;
            }
#pragma unroll
            for (int t = 0; t < 4; ++t)
#pragma unroll
                for (int r = 0; r < 8; ++r) oacc[u][t][r] *= csc[r];
        }

        // ---- reload P in A-fragment layout ---------------------------------
        asm volatile("s_wait_dscnt 0" ::: "memory");   // cross-lane RAW
        v16bf pfrag[2];
#pragma unroll
        for (int u = 0; u < 2; ++u) {
            const __bf16* prow = pbuf[wid][u] + mlane * 32;
#pragma unroll
            for (int i = 0; i < 8; ++i) {
                pfrag[u][i]     = prow[akb + i];
                pfrag[u][i + 8] = prow[8 + akb + i];
            }
        }

        // ---- O += P V (V fragments shared by both sub-tiles) ---------------
#pragma unroll
        for (int t = 0; t < 4; ++t) {
            const bf16_t* col =
                Vb + (size_t)(t * 16 + mlane) * SKVL + kv + hilo * 16;
            v16bf vf = ld_b_from_bf16(col);
#pragma unroll
            for (int u = 0; u < 2; ++u)
                oacc[u][t] = __builtin_amdgcn_wmma_f32_16x16x32_bf16(
                    false, pfrag[u], false, vf, (short)0, oacc[u][t],
                    false, false);
        }
    }

    // ---- epilogue: normalize, merge heads, write [Sq, B, E] fp32 -----------
#pragma unroll
    for (int u = 0; u < 2; ++u)
#pragma unroll
        for (int t = 0; t < 4; ++t) {
            const int d = t * 16 + mlane;
#pragma unroll
            for (int r = 0; r < 8; ++r) {
                const int s = q_base + u * 16 + r + hilo * 8;
                out[((size_t)s * BATCH + b) * EM + h * HEADD + d] =
                    oacc[u][t][r] / l[u][r];
            }
        }
}

// ---- host launcher ---------------------------------------------------------
extern "C" void kernel_launch(void* const* d_in, const int* in_sizes, int n_in,
                              void* d_out, int out_size, void* d_ws, size_t ws_size,
                              hipStream_t stream) {
    const float* q_em  = (const float*)d_in[0];
    const float* kv_em = (const float*)d_in[1];
    const float* Wq    = (const float*)d_in[2];
    const float* bq    = (const float*)d_in[3];
    const float* Wk    = (const float*)d_in[4];
    const float* bk    = (const float*)d_in[5];
    const float* Wv    = (const float*)d_in[6];
    const float* bv    = (const float*)d_in[7];
    float* out = (float*)d_out;

    const size_t nQ = (size_t)BATCH * NHEAD * SQL * HEADD;   // 8M bf16 elems
    bf16_t* Qw = (bf16_t*)d_ws;
    bf16_t* Kw = Qw + nQ;
    bf16_t* Vw = Kw + nQ;

    // 2048 wave-tiles per matrix / 8 waves per block = 256 blocks; y = {Q,K,V}
    dim3 gProj(256, 3, 1);
    mha_qkv_proj<<<gProj, 256, 0, stream>>>(q_em, kv_em, Wq, bq, Wk, bk, Wv, bv,
                                            Qw, Kw, Vw);

    // 4096 attention waves (32 q-rows each) / 8 = 512 blocks
    mha_flash_attn<<<512, 256, 0, stream>>>(Qw, Kw, Vw, out);
}
